// BiMamba4KT_27152783245621
// MI455X (gfx1250) — compile-verified, matrix operand order checked by reading, not run on hardware
//
#include <hip/hip_runtime.h>
#include <hip/hip_bf16.h>
#include <math.h>

// ---------------------------------------------------------------------------
// BiMamba knowledge-tracing model for gfx1250 (MI455X), wave32 + WMMA bf16
// + async global->LDS staging (ASYNCcnt path) in the selective scan.
// ---------------------------------------------------------------------------

#define BATCH   64
#define SEQ     256
#define DMODEL  256
#define DINNER  512
#define DSTATE  16
#define DTRANK  16
#define NTOK    (BATCH * SEQ)          // 16384 tokens
#define QUES    2048

typedef __attribute__((ext_vector_type(16))) __bf16 v16bf;
typedef __attribute__((ext_vector_type(8)))  float  v8f;

// Load one 16-lane-row bf16 WMMA fragment chunk (ISA 16-bit A/B layout):
// elems 0..7 -> K = kb0..kb0+7, elems 8..15 -> K = kb1..kb1+7.
__device__ __forceinline__ v16bf load_frag(const float* __restrict__ row,
                                           int kb0, int kb1)
{
    v16bf f;
#pragma unroll
    for (int j = 0; j < 8; ++j) {
        f[j]     = (__bf16)row[kb0 + j];
        f[j + 8] = (__bf16)row[kb1 + j];
    }
    return f;
}

__device__ __forceinline__ float apply_act(float v, int ACT)
{
    if (ACT == 1) v = 0.5f * v * (1.0f + erff(v * 0.70710678118654752f)); // GELU
    else if (ACT == 2) v = (v > 20.0f) ? v : log1pf(expf(v));             // softplus
    return v;
}

// ------------------- Register-blocked WMMA GEMM (32x64/wave) ---------------
// C[m,n] = act( sum_k A[m,k] * W[n,k] + bias[n] ).  Requires N%64==0, K%32==0.
// One wave computes a 32(M) x 64(N) tile: 2 A-fragments x 4 B-fragments ->
// 8 v_wmma_f32_16x16x32_bf16 per 32-wide K step.
template<int ACT>
__global__ __launch_bounds__(256)
void wmma_gemm_blk_kernel(const float* __restrict__ A, int lda,
                          const float* __restrict__ W,
                          const float* __restrict__ bias,
                          float* __restrict__ C, int ldc,
                          int M, int N, int K)
{
    const int tilesN = N >> 6;
    const int totalTiles = (M >> 5) * tilesN;
    const int wave = blockIdx.x * (blockDim.x >> 5) + (threadIdx.x >> 5);
    if (wave >= totalTiles) return;          // uniform per-wave exit

    const int lane = threadIdx.x & 31;
    const int m0 = (wave / tilesN) << 5;
    const int n0 = (wave % tilesN) << 6;
    const int half = lane >> 4;
    const int l    = lane & 15;

    const float* __restrict__ ar0 = A + (size_t)(m0 + l)      * (size_t)lda;
    const float* __restrict__ ar1 = A + (size_t)(m0 + 16 + l) * (size_t)lda;
    const float* __restrict__ br0 = W + (size_t)(n0 + l)      * (size_t)K;
    const float* __restrict__ br1 = W + (size_t)(n0 + 16 + l) * (size_t)K;
    const float* __restrict__ br2 = W + (size_t)(n0 + 32 + l) * (size_t)K;
    const float* __restrict__ br3 = W + (size_t)(n0 + 48 + l) * (size_t)K;

    v8f c00 = {}, c01 = {}, c02 = {}, c03 = {};
    v8f c10 = {}, c11 = {}, c12 = {}, c13 = {};

    for (int k0 = 0; k0 < K; k0 += 32) {
        const int kb0 = k0 + half * 8;
        const int kb1 = k0 + 16 + half * 8;
        // Hint the next K-slab toward the caches (global_prefetch_b8).
        __builtin_prefetch(ar0 + k0 + 32, 0, 1);
        __builtin_prefetch(br0 + k0 + 32, 0, 1);

        const v16bf a0 = load_frag(ar0, kb0, kb1);
        const v16bf a1 = load_frag(ar1, kb0, kb1);
        const v16bf b0 = load_frag(br0, kb0, kb1);
        const v16bf b1 = load_frag(br1, kb0, kb1);
        const v16bf b2 = load_frag(br2, kb0, kb1);
        const v16bf b3 = load_frag(br3, kb0, kb1);

        c00 = __builtin_amdgcn_wmma_f32_16x16x32_bf16(false, a0, false, b0, (short)0, c00, false, false);
        c01 = __builtin_amdgcn_wmma_f32_16x16x32_bf16(false, a0, false, b1, (short)0, c01, false, false);
        c02 = __builtin_amdgcn_wmma_f32_16x16x32_bf16(false, a0, false, b2, (short)0, c02, false, false);
        c03 = __builtin_amdgcn_wmma_f32_16x16x32_bf16(false, a0, false, b3, (short)0, c03, false, false);
        c10 = __builtin_amdgcn_wmma_f32_16x16x32_bf16(false, a1, false, b0, (short)0, c10, false, false);
        c11 = __builtin_amdgcn_wmma_f32_16x16x32_bf16(false, a1, false, b1, (short)0, c11, false, false);
        c12 = __builtin_amdgcn_wmma_f32_16x16x32_bf16(false, a1, false, b2, (short)0, c12, false, false);
        c13 = __builtin_amdgcn_wmma_f32_16x16x32_bf16(false, a1, false, b3, (short)0, c13, false, false);
    }

    auto epi = [&](const v8f& a, int mr, int n) {
        const float bv = bias ? bias[n] : 0.0f;
#pragma unroll
        for (int r = 0; r < 8; ++r) {
            C[(size_t)(mr + r) * (size_t)ldc + n] = apply_act(a[r] + bv, ACT);
        }
    };
    const int nb = n0 + l;
    const int mr0 = m0 + half * 8;           // lanes 16-31 hold rows M+8..M+15
    epi(c00, mr0,      nb);
    epi(c01, mr0,      nb + 16);
    epi(c02, mr0,      nb + 32);
    epi(c03, mr0,      nb + 48);
    epi(c10, mr0 + 16, nb);
    epi(c11, mr0 + 16, nb + 16);
    epi(c12, mr0 + 16, nb + 32);
    epi(c13, mr0 + 16, nb + 48);
}

// ------------------- Generic WMMA GEMM (16x16/wave) ------------------------
// Handles odd shapes (x_proj N=48, dt_proj K=16).  Branch-free main loop,
// single zero-padded tail chunk.
template<int ACT>
__global__ __launch_bounds__(256)
void wmma_gemm_kernel(const float* __restrict__ A, int lda,
                      const float* __restrict__ W,
                      const float* __restrict__ bias,
                      float* __restrict__ C, int ldc,
                      int M, int N, int K)
{
    const int tilesN = N >> 4;
    const int totalTiles = (M >> 4) * tilesN;
    const int wave = blockIdx.x * (blockDim.x >> 5) + (threadIdx.x >> 5);
    if (wave >= totalTiles) return;

    const int lane = threadIdx.x & 31;
    const int m0 = (wave / tilesN) << 4;
    const int n0 = (wave % tilesN) << 4;
    const int half = lane >> 4;
    const int l    = lane & 15;

    const float* __restrict__ arow = A + (size_t)(m0 + l) * (size_t)lda;
    const float* __restrict__ brow = W + (size_t)(n0 + l) * (size_t)K;

    v8f acc = {};
    const int kmain = K & ~31;

    for (int k0 = 0; k0 < kmain; k0 += 32) {
        const v16bf af  = load_frag(arow, k0 + half * 8, k0 + 16 + half * 8);
        const v16bf bfv = load_frag(brow, k0 + half * 8, k0 + 16 + half * 8);
        acc = __builtin_amdgcn_wmma_f32_16x16x32_bf16(
                  false, af, false, bfv, (short)0, acc, false, false);
    }
    if (kmain < K) {                          // zero-padded tail (dt_proj K=16)
        v16bf af, bfv;
        const int kb0 = kmain + half * 8;
        const int kb1 = kmain + 16 + half * 8;
#pragma unroll
        for (int j = 0; j < 8; ++j) {
            const int ka = kb0 + j, kc = kb1 + j;
            af[j]      = (__bf16)((ka < K) ? arow[ka] : 0.0f);
            af[j + 8]  = (__bf16)((kc < K) ? arow[kc] : 0.0f);
            bfv[j]     = (__bf16)((ka < K) ? brow[ka] : 0.0f);
            bfv[j + 8] = (__bf16)((kc < K) ? brow[kc] : 0.0f);
        }
        acc = __builtin_amdgcn_wmma_f32_16x16x32_bf16(
                  false, af, false, bfv, (short)0, acc, false, false);
    }

    const int   n  = n0 + l;
    const float bv = bias ? bias[n] : 0.0f;
    const int   mr = m0 + half * 8;
#pragma unroll
    for (int r = 0; r < 8; ++r) {
        C[(size_t)(mr + r) * (size_t)ldc + n] = apply_act(acc[r] + bv, ACT);
    }
}

// ------------------------------ LayerNorm ----------------------------------
// out = LN( A[tok] + cb * B2[tok'] ) over DMODEL, tok' optionally seq-flipped.
__global__ __launch_bounds__(256)
void ln_kernel(const float* __restrict__ A,
               const float* __restrict__ B2, float cb, int revB,
               const float* __restrict__ g, const float* __restrict__ beta,
               float eps, float* __restrict__ out)
{
    __shared__ float ssum[256];
    __shared__ float ssq[256];
    const int tok = blockIdx.x;
    const int f   = threadIdx.x;
    const size_t base = (size_t)tok * DMODEL + f;
    float x = A[base];
    if (B2) {
        const int b = tok / SEQ, t = tok % SEQ;
        const int tok2 = revB ? (b * SEQ + (SEQ - 1 - t)) : tok;
        x += cb * B2[(size_t)tok2 * DMODEL + f];
    }
    ssum[f] = x; ssq[f] = x * x;
    __syncthreads();
    for (int s = 128; s > 0; s >>= 1) {
        if (f < s) { ssum[f] += ssum[f + s]; ssq[f] += ssq[f + s]; }
        __syncthreads();
    }
    const float m = ssum[0] * (1.0f / DMODEL);
    const float v = fmaxf(ssq[0] * (1.0f / DMODEL) - m * m, 0.0f);
    out[base] = (x - m) * rsqrtf(v + eps) * g[f] + beta[f];
}

// ----------------------- Embedding gather + LayerNorm ----------------------
__global__ __launch_bounds__(256)
void embed_ln_kernel(const int* __restrict__ qa, const float* __restrict__ emb,
                     const float* __restrict__ g, const float* __restrict__ beta,
                     float* __restrict__ out)
{
    __shared__ float ssum[256];
    __shared__ float ssq[256];
    const int tok = blockIdx.x;
    const int f   = threadIdx.x;
    const float x = emb[(size_t)qa[tok] * DMODEL + f];
    ssum[f] = x; ssq[f] = x * x;
    __syncthreads();
    for (int s = 128; s > 0; s >>= 1) {
        if (f < s) { ssum[f] += ssum[f + s]; ssq[f] += ssq[f + s]; }
        __syncthreads();
    }
    const float m = ssum[0] * (1.0f / DMODEL);
    const float v = fmaxf(ssq[0] * (1.0f / DMODEL) - m * m, 0.0f);
    out[(size_t)tok * DMODEL + f] = (x - m) * rsqrtf(v + 1e-12f) * g[f] + beta[f];
}

// ------------------- Depthwise causal conv (k=4) + SiLU --------------------
// Reads first DINNER channels of xz (B,S,1024) in direction-aware ("virtual")
// time; writes xi (B,S,512) in virtual order.
__global__ __launch_bounds__(256)
void conv_silu_kernel(const float* __restrict__ xz,
                      const float* __restrict__ w,   // (512,4)
                      const float* __restrict__ cb,  // (512)
                      int rev, float* __restrict__ xi)
{
    const int idx = blockIdx.x * blockDim.x + threadIdx.x;
    if (idx >= NTOK * DINNER) return;
    const int d   = idx & (DINNER - 1);
    const int tok = idx >> 9;
    const int t   = tok % SEQ;           // virtual time
    const int b   = tok / SEQ;
    float acc = cb[d];
    const float* wd = w + d * 4;
#pragma unroll
    for (int k = 0; k < 4; ++k) {
        const int tp = t + k - 3;
        if (tp >= 0) {
            const int phys = rev ? (SEQ - 1 - tp) : tp;
            acc += wd[k] * xz[((size_t)(b * SEQ + phys) << 10) + d];
        }
    }
    acc = acc / (1.0f + __expf(-acc));   // SiLU
    xi[(size_t)tok * DINNER + d] = acc;
}

// --------------------------- Selective scan --------------------------------
// One thread per (batch, channel); 16-wide state kept in VGPRs.  Each block's
// 256 threads share one batch row, so the per-step B(16)/C(16) vectors are
// staged into LDS with gfx1250 async global->LDS copies (ASYNCcnt path,
// no VGPR round-trip): lanes 0..31 of wave 0 each deposit one dword, the
// issuing wave drains ASYNCcnt, then the workgroup barrier publishes the tile.
// Epilogue fuses the D-skip and silu(z) gate (z read in physical time for the
// bwd pass).
__global__ __launch_bounds__(256)
void scan_kernel(const float* __restrict__ xi,   const float* __restrict__ dt,
                 const float* __restrict__ xdbl, const float* __restrict__ xz,
                 const float* __restrict__ A_log, const float* __restrict__ Dskip,
                 int rev, float* __restrict__ yout)
{
    __shared__ float sBC[2 * DSTATE];
    const int idx = blockIdx.x * blockDim.x + threadIdx.x;
    if (idx >= BATCH * DINNER) return;
    const int d = idx & (DINNER - 1);
    const int b = idx >> 9;              // uniform across the block

    // Wave-relative LDS byte address for this lane's staging slot
    // (flat shared-pointer low 32 bits == LDS address, ISA 10.2 aperture rules).
    const unsigned int lds_addr =
        (unsigned int)(unsigned long long)(&sBC[threadIdx.x & 31]);

    float An[DSTATE];
#pragma unroll
    for (int n = 0; n < DSTATE; ++n) An[n] = -__expf(A_log[d * DSTATE + n]);
    const float Dd = Dskip[d];

    float h[DSTATE];
#pragma unroll
    for (int n = 0; n < DSTATE; ++n) h[n] = 0.0f;

    for (int t = 0; t < SEQ; ++t) {
        const size_t tok = (size_t)(b * SEQ + t);
        __syncthreads();                                  // sBC free to rewrite
        if (threadIdx.x < 2 * DSTATE) {
            const float* gsrc = xdbl + tok * 48 + DTRANK + threadIdx.x;
            asm volatile("global_load_async_to_lds_b32 %0, %1, off"
                         :: "v"(lds_addr), "v"(gsrc)
                         : "memory");
        }
        asm volatile("s_wait_asynccnt 0x0" ::: "memory"); // issuing wave drains
        __syncthreads();                                  // publish to block

        const float dtv = dt[tok * DINNER + d];
        const float u   = xi[tok * DINNER + d];
        const float du  = dtv * u;
        float y = 0.0f;
#pragma unroll
        for (int n = 0; n < DSTATE; ++n) {
            const float dA = __expf(dtv * An[n]);
            h[n] = dA * h[n] + du * sBC[n];
            y += h[n] * sBC[DSTATE + n];
        }
        const int phys = rev ? (SEQ - 1 - t) : t;
        const float zv = xz[((size_t)(b * SEQ + phys) << 10) + DINNER + d];
        const float sz = zv / (1.0f + __expf(-zv));
        yout[tok * DINNER + d] = (y + u * Dd) * sz;
    }
}

// ------------------------------- Host side ---------------------------------
static inline void launch_gemm(int act, const float* A, int lda, const float* W,
                               const float* bias, float* C, int ldc,
                               int N, int K, hipStream_t s)
{
    if ((N & 63) == 0 && (K & 31) == 0) {
        const int tiles  = (NTOK / 32) * (N / 64);
        const int blocks = (tiles + 7) / 8;
        if (act == 0)
            wmma_gemm_blk_kernel<0><<<blocks, 256, 0, s>>>(A, lda, W, bias, C, ldc, NTOK, N, K);
        else
            wmma_gemm_blk_kernel<1><<<blocks, 256, 0, s>>>(A, lda, W, bias, C, ldc, NTOK, N, K);
    } else {
        const int tiles  = (NTOK / 16) * (N / 16);
        const int blocks = (tiles + 7) / 8;
        if (act == 2)
            wmma_gemm_kernel<2><<<blocks, 256, 0, s>>>(A, lda, W, bias, C, ldc, NTOK, N, K);
        else
            wmma_gemm_kernel<0><<<blocks, 256, 0, s>>>(A, lda, W, bias, C, ldc, NTOK, N, K);
    }
}

// Workspace layout (floats); larger buffers are reused across phases.
#define O_H    ((size_t)0)
#define O_XN   (O_H    + (size_t)NTOK * DMODEL)
#define O_XZ   (O_XN   + (size_t)NTOK * DMODEL)
#define O_XI   (O_XZ   + (size_t)NTOK * 2 * DINNER)
#define O_DT   (O_XI   + (size_t)NTOK * DINNER)
#define O_Y    (O_DT   + (size_t)NTOK * DINNER)
#define O_XDBL (O_Y    + (size_t)NTOK * DINNER)
#define O_OF   (O_XDBL + (size_t)NTOK * 48)
#define O_OB   (O_OF   + (size_t)NTOK * DMODEL)
// aliases: m -> O_XN, t1/t2 -> O_XZ, ff -> O_XI, h2 -> O_DT, hs -> O_Y

extern "C" void kernel_launch(void* const* d_in, const int* in_sizes, int n_in,
                              void* d_out, int out_size, void* d_ws, size_t ws_size,
                              hipStream_t stream)
{
    (void)in_sizes; (void)n_in; (void)out_size; (void)ws_size;

    const int*   qa      = (const int*)  d_in[0];
    const float* emb     = (const float*)d_in[2];
    const float* ln0_g   = (const float*)d_in[3];
    const float* ln0_b   = (const float*)d_in[4];
    const float* in_w    = (const float*)d_in[5];
    const float* conv_w  = (const float*)d_in[6];
    const float* conv_b  = (const float*)d_in[7];
    const float* xproj_w = (const float*)d_in[8];
    const float* dt_w    = (const float*)d_in[9];
    const float* dt_b    = (const float*)d_in[10];
    const float* A_log   = (const float*)d_in[11];
    const float* Dskip   = (const float*)d_in[12];
    const float* out_w   = (const float*)d_in[13];
    const float* n1_g    = (const float*)d_in[14];
    const float* n1_b    = (const float*)d_in[15];
    const float* n2_g    = (const float*)d_in[16];
    const float* n2_b    = (const float*)d_in[17];
    const float* lln_g   = (const float*)d_in[18];
    const float* lln_b   = (const float*)d_in[19];
    const float* fln_g   = (const float*)d_in[20];
    const float* fln_b   = (const float*)d_in[21];
    const float* bff1_w  = (const float*)d_in[22];
    const float* bff1_b  = (const float*)d_in[23];
    const float* bff2_w  = (const float*)d_in[24];
    const float* bff2_b  = (const float*)d_in[25];
    const float* ffn1_w  = (const float*)d_in[26];
    const float* ffn1_b  = (const float*)d_in[27];
    const float* ffn2_w  = (const float*)d_in[28];
    const float* ffn2_b  = (const float*)d_in[29];
    const float* fc_w    = (const float*)d_in[30];
    const float* fc_b    = (const float*)d_in[31];

    float* ws   = (float*)d_ws;
    float* h    = ws + O_H;
    float* xn   = ws + O_XN;     // also "m" after bidirectional merge
    float* xz   = ws + O_XZ;     // also t1/t2 (MLP hidden, same footprint)
    float* xi   = ws + O_XI;     // also "ff"
    float* dtb  = ws + O_DT;     // also "h2"
    float* y    = ws + O_Y;      // also "hs"
    float* xdbl = ws + O_XDBL;
    float* of   = ws + O_OF;
    float* ob   = ws + O_OB;
    float* out  = (float*)d_out;

    // Embedding gather + LN0 (eps 1e-12)
    embed_ln_kernel<<<NTOK, 256, 0, stream>>>(qa, emb, ln0_g, ln0_b, h);

    for (int L = 0; L < 2; ++L) {
        // xn = LN(h) (eps 1e-5)
        ln_kernel<<<NTOK, 256, 0, stream>>>(h, nullptr, 0.0f, 0,
                                            n1_g + L * DMODEL, n1_b + L * DMODEL,
                                            1e-5f, xn);
        // xz = xn @ in_w^T  (256 -> 1024), shared by both directions
        launch_gemm(0, xn, DMODEL, in_w + (size_t)L * 2 * DINNER * DMODEL,
                    nullptr, xz, 2 * DINNER, 2 * DINNER, DMODEL, stream);

        for (int dir = 0; dir < 2; ++dir) {
            conv_silu_kernel<<<(NTOK * DINNER) / 256, 256, 0, stream>>>(
                xz, conv_w + (size_t)L * DINNER * 4, conv_b + (size_t)L * DINNER,
                dir, xi);
            // x_dbl = xi @ xproj^T (512 -> 48)
            launch_gemm(0, xi, DINNER, xproj_w + (size_t)L * 48 * DINNER,
                        nullptr, xdbl, 48, 48, DINNER, stream);
            // dt = softplus(x_dbl[:, :16] @ dt_w^T + dt_b) (16 -> 512)
            launch_gemm(2, xdbl, 48, dt_w + (size_t)L * DINNER * DTRANK,
                        dt_b + (size_t)L * DINNER, dtb, DINNER, DINNER, DTRANK,
                        stream);
            scan_kernel<<<(BATCH * DINNER) / 256, 256, 0, stream>>>(
                xi, dtb, xdbl, xz,
                A_log + (size_t)L * DINNER * DSTATE, Dskip + (size_t)L * DINNER,
                dir, y);
            // out_dir = y @ out_w^T (512 -> 256)
            launch_gemm(0, y, DINNER, out_w + (size_t)L * DMODEL * DINNER,
                        nullptr, dir ? ob : of, DMODEL, DMODEL, DINNER, stream);
        }

        // m = LN(fwd + flip(bwd)) (eps 1e-5)  -> xn buffer
        ln_kernel<<<NTOK, 256, 0, stream>>>(of, ob, 1.0f, 1,
                                            n2_g + L * DMODEL, n2_b + L * DMODEL,
                                            1e-5f, xn);
        // ff = GELU(m @ bff1^T + b) @ bff2^T + b
        launch_gemm(1, xn, DMODEL, bff1_w + (size_t)L * 4 * DMODEL * DMODEL,
                    bff1_b + (size_t)L * 4 * DMODEL, xz, 4 * DMODEL,
                    4 * DMODEL, DMODEL, stream);
        launch_gemm(0, xz, 4 * DMODEL, bff2_w + (size_t)L * DMODEL * 4 * DMODEL,
                    bff2_b + (size_t)L * DMODEL, xi, DMODEL,
                    DMODEL, 4 * DMODEL, stream);
        // h2 = LN(ff + 2h)  (bi_out + h == ff + 2h), eps 1e-12 -> dtb buffer
        ln_kernel<<<NTOK, 256, 0, stream>>>(xi, h, 2.0f, 0,
                                            lln_g + L * DMODEL, lln_b + L * DMODEL,
                                            1e-12f, dtb);
        // hs = GELU(h2 @ ffn1^T + b) @ ffn2^T + b
        launch_gemm(1, dtb, DMODEL, ffn1_w + (size_t)L * 4 * DMODEL * DMODEL,
                    ffn1_b + (size_t)L * 4 * DMODEL, xz, 4 * DMODEL,
                    4 * DMODEL, DMODEL, stream);
        launch_gemm(0, xz, 4 * DMODEL, ffn2_w + (size_t)L * DMODEL * 4 * DMODEL,
                    ffn2_b + (size_t)L * DMODEL, y, DMODEL,
                    DMODEL, 4 * DMODEL, stream);
        // h = LN(hs + h2) (eps 1e-12)
        ln_kernel<<<NTOK, 256, 0, stream>>>(y, dtb, 1.0f, 0,
                                            fln_g + L * DMODEL, fln_b + L * DMODEL,
                                            1e-12f, h);
    }

    // logits = h @ fc_w^T + fc_b  (256 -> 2048)
    launch_gemm(0, h, DMODEL, fc_w, fc_b, out, QUES, QUES, DMODEL, stream);
}
